// Generator_18176301597612
// MI455X (gfx1250) — matrix operand order repaired and checked by reading, not compiled, not run on hardware
//
#include <hip/hip_runtime.h>
#include <hip/hip_bf16.h>

// ---------------------------------------------------------------------------
// Sizes from the reference
// ---------------------------------------------------------------------------
#define BATCH 32
#define TWIN  64
#define NODE  512
#define HID   256
#define G4    1024          // 4*HID

typedef __attribute__((ext_vector_type(16))) __bf16 v16bf;
typedef __attribute__((ext_vector_type(8)))  float  v8f;

#define DEVINL __device__ __forceinline__

DEVINL __bf16 f2bf(float x) { return (__bf16)x; }

DEVINL v8f wmma_bf16(v16bf a, v16bf b, v8f c) {
  // D = A(16x32) * B(32x16) + C, f32 accumulate
  return __builtin_amdgcn_wmma_f32_16x16x32_bf16(
      /*neg_a=*/false, a, /*neg_b=*/false, b,
      /*c_mod=*/(short)0, c, /*reuse_a=*/false, /*reuse_b=*/false);
}

DEVINL float sigmoidf_(float x) { return 1.f / (1.f + __expf(-x)); }
DEVINL float tanhf_(float x)    { float e = __expf(2.f * x); return (e - 1.f) / (e + 1.f); }

// ---------------------------------------------------------------------------
// Kernel 0: pack W_hh (1024x256 f32) -> bf16, same row-major layout.
// ---------------------------------------------------------------------------
__global__ __launch_bounds__(256) void k_pack_whh(const float* __restrict__ Whh,
                                                  __bf16* __restrict__ Wp) {
  int i = blockIdx.x * 256 + threadIdx.x;
  if (i < G4 * HID) Wp[i] = f2bf(Whh[i]);
}

// ---------------------------------------------------------------------------
// Kernel 1: XG[t*32+b][n] = x[b,t,:] @ W_ih^T + (b_ih+b_hh)   (parallel in t,b)
// M = B*T = 2048 rows (r = b*64+t), N = 1024, K = 512.  One wave per 16x16 tile.
// ---------------------------------------------------------------------------
__global__ __launch_bounds__(256) void k_xgates(const float* __restrict__ x,
                                                const float* __restrict__ Wih,
                                                const float* __restrict__ bih,
                                                const float* __restrict__ bhh,
                                                float* __restrict__ XG) {
  const int lane = threadIdx.x & 31;
  const int wave = threadIdx.x >> 5;
  const int task = blockIdx.x * 8 + wave;   // 8192 wave-tasks
  const int mt = task >> 6;                 // 0..127
  const int nt = task & 63;                 // 0..63
  const int r0 = mt * 16;
  const int n0 = nt * 16;
  const int lm = lane & 15;
  const bool hi = lane >= 16;

  const int ncol = n0 + lm;
  const float bias = bih[ncol] + bhh[ncol];
  v8f acc;
#pragma unroll
  for (int v = 0; v < 8; ++v) acc[v] = bias;

  const float* xrow = x + (r0 + lm) * NODE;     // A: row = batch-time row
  const float* wrow = Wih + ncol * NODE;        // B: column n = W_ih row (transposed)

#pragma unroll 2
  for (int kc = 0; kc < NODE / 32; ++kc) {
    v16bf a, b;
    const int ka = kc * 32 + (hi ? 16 : 0);
#pragma unroll
    for (int u = 0; u < 8; ++u) {
      a[u]     = f2bf(xrow[ka + u]);
      a[u + 8] = f2bf(xrow[ka + 8 + u]);
    }
    const int kb = kc * 32 + (hi ? 16 : 0);
#pragma unroll
    for (int u = 0; u < 16; ++u) b[u] = f2bf(wrow[kb + u]);
    acc = wmma_bf16(a, b, acc);
  }
  // D layout: lane holds N = ncol, rows M = v + 8*hi.  Write t-major for scan.
#pragma unroll
  for (int v = 0; v < 8; ++v) {
    const int m  = r0 + v + (hi ? 8 : 0);   // = b*64 + t
    const int bb = m >> 6, tt = m & 63;
    XG[(tt * BATCH + bb) * G4 + ncol] = acc[v];
  }
}

// ---------------------------------------------------------------------------
// Kernel 2: sequential LSTM scan. One workgroup, 32 waves.
// Wave (bt, jt) owns output tile: batches [bt*16,+16), hidden cols [jt*16,+16).
// c lives in registers, h is exchanged through LDS as bf16.
// ---------------------------------------------------------------------------
__global__ __launch_bounds__(1024) void k_scan(const float* __restrict__ XG,
                                               const __bf16* __restrict__ Wp,
                                               float* __restrict__ hn) {
  __shared__ __bf16 hbuf[BATCH][HID];    // 16 KB
  const int lane = threadIdx.x & 31;
  const int wave = threadIdx.x >> 5;
  const int lm = lane & 15;
  const bool hi = lane >= 16;
  const int bt = wave & 1;
  const int jt = wave >> 1;
  const int m0 = bt * 16;
  const int jcol = jt * 16 + lm;

  for (int i = threadIdx.x; i < BATCH * HID; i += 1024)
    (&hbuf[0][0])[i] = (__bf16)0.f;
  __syncthreads();

  float cst[8], hv[8];
#pragma unroll
  for (int v = 0; v < 8; ++v) { cst[v] = 0.f; hv[v] = 0.f; }

#pragma unroll 1
  for (int t = 0; t < TWIN; ++t) {
    v8f acc[4];
    // C init = precomputed x-part of the gates
#pragma unroll
    for (int g = 0; g < 4; ++g) {
      const int n = g * HID + jcol;
#pragma unroll
      for (int v = 0; v < 8; ++v) {
        const int m = m0 + v + (hi ? 8 : 0);
        acc[g][v] = XG[(t * BATCH + m) * G4 + n];
      }
    }
    // gates += h @ W_hh^T  (K = 256 -> 8 bf16 WMMA per gate tile)
#pragma unroll 2
    for (int kc = 0; kc < HID / 32; ++kc) {
      v16bf a;
      const int ka = kc * 32 + (hi ? 16 : 0);
      const __bf16* hr = &hbuf[m0 + lm][0];
#pragma unroll
      for (int u = 0; u < 8; ++u) {
        a[u]     = hr[ka + u];
        a[u + 8] = hr[ka + 8 + u];
      }
      const int kb = kc * 32 + (hi ? 16 : 0);
#pragma unroll
      for (int g = 0; g < 4; ++g) {
        v16bf b;
        const __bf16* wr = Wp + (g * HID + jcol) * HID + kb;
#pragma unroll
        for (int u = 0; u < 16; ++u) b[u] = wr[u];
        acc[g] = wmma_bf16(a, b, acc[g]);
      }
    }
    // element-wise LSTM cell update (c kept in VGPRs)
#pragma unroll
    for (int v = 0; v < 8; ++v) {
      const float ig = sigmoidf_(acc[0][v]);
      const float fg = sigmoidf_(acc[1][v]);
      const float gg = tanhf_(acc[2][v]);
      const float og = sigmoidf_(acc[3][v]);
      const float cc = fg * cst[v] + ig * gg;
      cst[v] = cc;
      hv[v] = og * tanhf_(cc);
    }
    __syncthreads();   // all waves done reading old h
#pragma unroll
    for (int v = 0; v < 8; ++v)
      hbuf[m0 + v + (hi ? 8 : 0)][jcol] = f2bf(hv[v]);
    __syncthreads();   // new h visible to all waves
  }
#pragma unroll
  for (int v = 0; v < 8; ++v)
    hn[(m0 + v + (hi ? 8 : 0)) * HID + jcol] = hv[v];
}

// ---------------------------------------------------------------------------
// Kernel 3: head GEMM + fused transpose mask.
// Each workgroup owns a transpose-pair of 16x16 (i,j)-blocks (I<=J, 528 tasks),
// computes a[b, Iblk, Jblk] and a[b, Jblk, Iblk] for all 32 batches into LDS,
// then masks and writes final output (single pass over f1_w: ~256 MB).
// WMMA roles: A = f1_w rows (16 n-rows x 32 k, f32->bf16 streamed from HBM),
//             B = hn^T tile (staged once in LDS as bf16).
// ---------------------------------------------------------------------------
__global__ __launch_bounds__(256) void k_head(const float* __restrict__ hn,
                                              const float* __restrict__ f1w,
                                              const float* __restrict__ f1b,
                                              float* __restrict__ out) {
  __shared__ float  ablk[2 * BATCH * 16 * 16];   // 64 KB raw results
  __shared__ __bf16 hnb[BATCH][HID];             // 16 KB staged hn (bf16)
  const int lane = threadIdx.x & 31;
  const int wave = threadIdx.x >> 5;
  const int lm = lane & 15;
  const bool hi = lane >= 16;

  // stage hn once (32x256 f32 -> bf16), row-major
  for (int i = threadIdx.x; i < BATCH * HID; i += 256)
    (&hnb[0][0])[i] = f2bf(hn[i]);
  __syncthreads();

  // unrank blockIdx.x -> (I, J) with I <= J over 32x32 blocks (528 pairs)
  int idx = blockIdx.x;
  int I = 0;
  while (idx >= (32 - I)) { idx -= (32 - I); ++I; }
  const int J = I + idx;

  // 64 combos: {blk} x {i2 row-in-block} x {batch half}
  for (int combo = wave; combo < 64; combo += 8) {
    const int blk = combo >> 5;
    const int rem = combo & 31;
    const int i2 = rem >> 1;
    const int btile = rem & 1;
    if (blk == 1 && I == J) continue;
    const int bi = blk ? J : I;
    const int bj = blk ? I : J;
    const int nbase = (bi * 16 + i2) * NODE + bj * 16;

    v8f acc;
#pragma unroll
    for (int v = 0; v < 8; ++v) acc[v] = f1b[nbase + v + (hi ? 8 : 0)];

    const float*  arow = f1w + (nbase + lm) * HID;   // 16 contiguous n-rows
    const __bf16* brow = &hnb[btile * 16 + lm][0];   // B column = batch (LDS)
    // prefetch this wave's next combo's f1_w rows (speculative, RT)
    __builtin_prefetch(f1w + (nbase + 4 * NODE + lm) * HID, 0, 0);

#pragma unroll 2
    for (int kc = 0; kc < HID / 32; ++kc) {
      v16bf a, b;
      const int ka = kc * 32 + (hi ? 16 : 0);
#pragma unroll
      for (int u = 0; u < 8; ++u) {
        a[u]     = f2bf(arow[ka + u]);
        a[u + 8] = f2bf(arow[ka + 8 + u]);
      }
      const int kb = kc * 32 + (hi ? 16 : 0);
#pragma unroll
      for (int u = 0; u < 16; ++u) b[u] = brow[kb + u];
      acc = wmma_bf16(a, b, acc);
    }
    // D: lane col = batch (btile*16+lm), row = j2 = v + 8*hi
    const int b = btile * 16 + lm;
    float* dst = &ablk[((blk * BATCH + b) * 16 + i2) * 16];
#pragma unroll
    for (int v = 0; v < 8; ++v) dst[v + (hi ? 8 : 0)] = acc[v];
  }
  __syncthreads();

  // mask phase: a[b,i,j] survives iff it strictly beats a[b,j,i]
  // (upper triangle wins ties, diagonal dies) and is > 0.
  for (int e = threadIdx.x; e < 2 * BATCH * 256; e += 256) {
    const int blk = e >> 13;
    if (blk == 1 && I == J) continue;
    const int rem = e & 8191;
    const int b = rem >> 8;
    const int i2 = (rem >> 4) & 15;
    const int j2 = rem & 15;
    const int bi = blk ? J : I;
    const int bj = blk ? I : J;
    const float val = ablk[((blk * BATCH + b) * 16 + i2) * 16 + j2];
    const int oblk = (I == J) ? blk : (1 - blk);
    const float tval = ablk[((oblk * BATCH + b) * 16 + j2) * 16 + i2];
    const int gi = bi * 16 + i2, gj = bj * 16 + j2;
    const bool wins = (val > tval) || ((val == tval) && (gi < gj));
    out[b * (NODE * NODE) + gi * NODE + gj] = (wins && val > 0.f) ? val : 0.f;
  }
}

// ---------------------------------------------------------------------------
// Launch: pack_whh & xgates (independent) -> scan -> head, all on `stream`.
// Workspace: XG 8 MB | W_hh bf16 512 KB | hn 32 KB  (~8.6 MB total)
// ---------------------------------------------------------------------------
extern "C" void kernel_launch(void* const* d_in, const int* in_sizes, int n_in,
                              void* d_out, int out_size, void* d_ws, size_t ws_size,
                              hipStream_t stream) {
  (void)in_sizes; (void)n_in; (void)out_size; (void)ws_size;
  // setup_inputs order: 0 in_shots (unused by reference), 1 in_node, 2 W_ih,
  // 3 W_hh, 4 b_ih, 5 b_hh, 6 f1_w, 7 f1_b
  const float* in_node = (const float*)d_in[1];
  const float* Wih = (const float*)d_in[2];
  const float* Whh = (const float*)d_in[3];
  const float* bih = (const float*)d_in[4];
  const float* bhh = (const float*)d_in[5];
  const float* f1w = (const float*)d_in[6];
  const float* f1b = (const float*)d_in[7];
  float* out = (float*)d_out;

  char* ws = (char*)d_ws;
  float*  XG = (float*)ws;                                   // 64*32*1024*4 = 8 MB
  __bf16* Wp = (__bf16*)(ws + 8u * 1024u * 1024u);           // 512 KB
  float*  hn = (float*)(ws + 8u * 1024u * 1024u + 512u * 1024u); // 32 KB

  k_pack_whh<<<G4 * HID / 256, 256, 0, stream>>>(Whh, Wp);
  k_xgates<<<1024, 256, 0, stream>>>(in_node, Wih, bih, bhh, XG);
  k_scan<<<1, 1024, 0, stream>>>(XG, Wp, hn);
  k_head<<<528, 256, 0, stream>>>(hn, f1w, f1b, out);
}
